// CausalSelfAttention_33792802685116
// MI455X (gfx1250) — matrix operand order repaired and checked by reading, not compiled, not run on hardware
//
#include <hip/hip_runtime.h>

#define BATCH 2
#define SEQ   2048
#define DIMN  2048
#define NH    16
#define NKV   4
#define HDIM  128
#define KVD   512
#define GRP   4
#define LRANK 16

typedef __attribute__((ext_vector_type(16))) __bf16 v16bf;
typedef __attribute__((ext_vector_type(8)))  __bf16 v8bf;
typedef __attribute__((ext_vector_type(8)))  float  v8f;
typedef __attribute__((ext_vector_type(4)))  float  v4f;

__device__ __forceinline__ __bf16 f2bf(float f) {
  unsigned u = __builtin_bit_cast(unsigned, f);
  unsigned r = (u + 0x7FFFu + ((u >> 16) & 1u)) >> 16;
  unsigned short s = (unsigned short)r;
  return __builtin_bit_cast(__bf16, s);
}

__device__ __forceinline__ v16bf cat16(v8bf lo, v8bf hi) {
  return __builtin_shufflevector(lo, hi, 0, 1, 2, 3, 4, 5, 6, 7, 8, 9, 10, 11, 12, 13, 14, 15);
}
// A-fragment: two contiguous 8-elem chunks at +0 and +16 (ISA 16-bit A layout)
__device__ __forceinline__ v16bf ldfragA(const __bf16* base) {
  v8bf lo = *(const v8bf*)(base);
  v8bf hi = *(const v8bf*)(base + 16);
  return cat16(lo, hi);
}
// B-fragment: two contiguous 8-elem chunks at +0 and +16 (column-major operand)
__device__ __forceinline__ v16bf ldfrag16(const __bf16* base) {
  v8bf lo = *(const v8bf*)(base);
  v8bf hi = *(const v8bf*)(base + 16);
  return cat16(lo, hi);
}

// ---------------- W_eff^T = (W + A @ B)^T  -> bf16, stored N x K ----------------
__global__ void build_weight_t(const float* __restrict__ W, const float* __restrict__ A,
                               const float* __restrict__ Bm, __bf16* __restrict__ outT,
                               int D, int N) {
  int idx = blockIdx.x * blockDim.x + threadIdx.x;
  if (idx >= D * N) return;
  int n = idx / D, d = idx % D;  // d fastest -> contiguous transposed writes
  float acc = W[(size_t)d * N + n];
#pragma unroll
  for (int r = 0; r < LRANK; ++r) acc += A[d * LRANK + r] * Bm[r * N + n];
  outT[(size_t)n * D + d] = f2bf(acc);
}

__global__ void cvt_bf16(const float* __restrict__ in, __bf16* __restrict__ out, int n) {
  int i = blockIdx.x * blockDim.x + threadIdx.x;
  if (i < n) out[i] = f2bf(in[i]);
}

// V: (b,t,kh,hd) f32 -> (b,kh,hd,t) bf16 (transposed for contiguous P@V B-fragments)
__global__ void cvt_v_t(const float* __restrict__ Vf, __bf16* __restrict__ Vt) {
  int idx = blockIdx.x * blockDim.x + threadIdx.x;
  if (idx >= BATCH * SEQ * KVD) return;
  int d = idx % KVD;
  int t = (idx / KVD) % SEQ;
  int b = idx / (KVD * SEQ);
  int kh = d / HDIM, hd = d % HDIM;
  Vt[(((size_t)b * NKV + kh) * HDIM + hd) * SEQ + t] = f2bf(Vf[idx]);
}

// ---------------- bf16 GEMM: C[M,N] = A[M,K] @ Bt[N,K]^T ----------------
// one wave computes a 64x32 tile (4x2 WMMA accumulators)
#define MW 4
#define NW 2
__global__ void gemm_bf16(const __bf16* __restrict__ A, const __bf16* __restrict__ Bt,
                          float* __restrict__ C, int M, int N, int K) {
  int wave = (blockIdx.x * blockDim.x + threadIdx.x) >> 5;
  int lane = threadIdx.x & 31;
  int ln = lane & 15;
  bool hi = lane >= 16;
  int nWaveTiles = N / (16 * NW);
  int mw = wave / nWaveTiles, nw = wave % nWaveTiles;
  int mBase = mw * 16 * MW, nBase = nw * 16 * NW;

  const __bf16* aRow[MW];
  const __bf16* bCol[NW];
#pragma unroll
  for (int i = 0; i < MW; ++i) aRow[i] = A + (size_t)(mBase + i * 16 + ln) * K + (hi ? 8 : 0);
#pragma unroll
  for (int j = 0; j < NW; ++j) bCol[j] = Bt + (size_t)(nBase + j * 16 + ln) * K + (hi ? 8 : 0);

  v8f acc[MW][NW];
  v8f vz = {};
#pragma unroll
  for (int i = 0; i < MW; ++i)
#pragma unroll
    for (int j = 0; j < NW; ++j) acc[i][j] = vz;

  for (int k0 = 0; k0 < K; k0 += 32) {
    if (k0 + 32 < K) {
#pragma unroll
      for (int i = 0; i < MW; ++i) __builtin_prefetch(aRow[i] + k0 + 32, 0, 3);
#pragma unroll
      for (int j = 0; j < NW; ++j) __builtin_prefetch(bCol[j] + k0 + 32, 0, 3);
    }
    v16bf af[MW], bfr[NW];
#pragma unroll
    for (int i = 0; i < MW; ++i) af[i] = ldfragA(aRow[i] + k0);
#pragma unroll
    for (int j = 0; j < NW; ++j) bfr[j] = ldfrag16(bCol[j] + k0);
#pragma unroll
    for (int i = 0; i < MW; ++i)
#pragma unroll
      for (int j = 0; j < NW; ++j)
        acc[i][j] = __builtin_amdgcn_wmma_f32_16x16x32_bf16(false, af[i], false, bfr[j],
                                                            (short)0, acc[i][j], false, false);
  }
#pragma unroll
  for (int i = 0; i < MW; ++i)
#pragma unroll
    for (int j = 0; j < NW; ++j)
#pragma unroll
      for (int r = 0; r < 8; ++r) {
        int m = mBase + i * 16 + r + (hi ? 8 : 0);
        C[(size_t)m * N + nBase + j * 16 + ln] = acc[i][j][r];
      }
}

// ---------------- per-head RMSNorm + RoPE (+gain) -> bf16 ----------------
__global__ void qk_postprocess(const float* __restrict__ X, __bf16* __restrict__ Y,
                               const float* __restrict__ gain, int nh) {
  int t = blockIdx.x, h = blockIdx.y, b = blockIdx.z;
  int d = threadIdx.x;  // 0..127
  int rowDim = nh * HDIM;
  const float* xp = X + ((size_t)b * SEQ + t) * rowDim + h * HDIM;
  __shared__ float buf[HDIM];
  __shared__ float red[4];
  float v = xp[d];
  float ss = v * v;
  for (int m = 16; m; m >>= 1) ss += __shfl_xor(ss, m, 32);
  if ((d & 31) == 0) red[d >> 5] = ss;
  __syncthreads();
  float scale = rsqrtf((red[0] + red[1] + red[2] + red[3]) * (1.0f / HDIM) + 1.1920929e-7f);
  float nv = v * scale;
  buf[d] = nv;
  __syncthreads();
  float outv;
  if (d < 64) {
    int j = (d < 32) ? d : d - 32;
    float x1 = buf[j], x2 = buf[j + 32];
    float inv_freq = __expf(-(float)j * (9.210340371976184f / 32.0f));  // 10000^(-2j/64)
    float ang = (float)t * inv_freq;
    float sn, cs;
    __sincosf(ang, &sn, &cs);
    outv = (d < 32) ? (x1 * cs + x2 * sn) : (-x1 * sn + x2 * cs);
  } else {
    outv = nv;
  }
  if (gain) outv *= gain[h];
  Y[((size_t)b * SEQ + t) * rowDim + h * HDIM + d] = f2bf(outv);
}

// ---- async stage of one 32-key K tile (32x128) and V^T tile (128x32) into LDS ----
__device__ __forceinline__ void async_stage_kv(const __bf16* __restrict__ KpH,
                                               const __bf16* __restrict__ VpH,
                                               int kBase, __bf16* kb, __bf16* vb, int lane) {
  int rowk = lane >> 4;    // 0..1
  int col16 = lane & 15;   // 16B chunk within a 256B K row
#pragma unroll
  for (int c = 0; c < 16; ++c) {
    int row = c * 2 + rowk;  // key row 0..31
    const __bf16* g = KpH + (size_t)(kBase + row) * KVD + col16 * 8;
    unsigned l = (unsigned)(uintptr_t)(kb + row * HDIM + col16 * 8);
    asm volatile("global_load_async_to_lds_b128 %0, %1, off" ::"v"(l), "v"(g) : "memory");
  }
#pragma unroll
  for (int c = 0; c < 16; ++c) {
    int idx = c * 32 + lane;
    int row = idx >> 2;  // hd 0..127
    int ch = idx & 3;    // 16B chunk within a 64B V^T row
    const __bf16* g = VpH + (size_t)row * SEQ + kBase + ch * 8;
    unsigned l = (unsigned)(uintptr_t)(vb + row * 32 + ch * 8);
    asm volatile("global_load_async_to_lds_b128 %0, %1, off" ::"v"(l), "v"(g) : "memory");
  }
}

// ---------------- flash attention: one wave per (b, head, 16-query tile) ----------------
// K/V tiles double-buffered in LDS via GLOBAL_LOAD_ASYNC_TO_LDS (ASYNCcnt pipelining)
__global__ void attn_fwd(const __bf16* __restrict__ Q, const __bf16* __restrict__ Kt,
                         const __bf16* __restrict__ Vt, float* __restrict__ Y) {
  int qTile = blockIdx.x, h = blockIdx.y, b = blockIdx.z;
  int kh = h / GRP;
  int lane = threadIdx.x;
  int ln = lane & 15;
  bool hi = lane >= 16;
  int qBase = qTile * 16;

  const __bf16* Qp = Q + (size_t)b * SEQ * DIMN + (size_t)h * HDIM;
  const __bf16* KpH = Kt + (size_t)b * SEQ * KVD + (size_t)kh * HDIM;       // (t, hd)
  const __bf16* VpH = Vt + ((size_t)b * NKV + kh) * HDIM * SEQ;             // (hd, t)

  __shared__ __bf16 kbuf[2][32 * HDIM];  // 2 x 8KB
  __shared__ __bf16 vbuf[2][HDIM * 32];  // 2 x 8KB
  __shared__ float pbuf[16 * 32];        // 2KB

  // Q fragments: 4 chunks of K=32 over HD=128 (vectorized b128 loads)
  v16bf qf[4];
  {
    const __bf16* qr = Qp + (size_t)(qBase + ln) * DIMN + (hi ? 8 : 0);
#pragma unroll
    for (int c = 0; c < 4; ++c) qf[c] = ldfragA(qr + c * 32);
  }

  float mrun[8], lrun[8];
  v8f o[8];
  v8f vz = {};
#pragma unroll
  for (int r = 0; r < 8; ++r) { mrun[r] = -3.0e38f; lrun[r] = 0.0f; o[r] = vz; }

  const float scale = 0.08838834764831845f;  // 128^-0.5
  int nKB = (qBase + 16 + 31) / 32;

  // preload block 0
  async_stage_kv(KpH, VpH, 0, kbuf[0], vbuf[0], lane);

  for (int j = 0; j < nKB; ++j) {
    int kBase = j * 32;
    const __bf16* kb = kbuf[j & 1];
    const __bf16* vb = vbuf[j & 1];
    if (j + 1 < nKB) {
      // drain any ds reads of the buffer we are about to overwrite, then stream next tile
      asm volatile("s_wait_dscnt 0x0" ::: "memory");
      async_stage_kv(KpH, VpH, kBase + 32, kbuf[(j + 1) & 1], vbuf[(j + 1) & 1], lane);
      // async loads complete in order: <=32 outstanding means current tile has landed
      asm volatile("s_wait_asynccnt 0x20" ::: "memory");
    } else {
      asm volatile("s_wait_asynccnt 0x0" ::: "memory");
    }
    // S = Q @ K^T : two 16x16 key sub-tiles, accumulate over 4 K-chunks (from LDS)
    v8f s0 = vz, s1 = vz;
#pragma unroll
    for (int c = 0; c < 4; ++c) {
      v16bf b0 = ldfrag16(kb + (size_t)ln * HDIM + c * 32 + (hi ? 8 : 0));
      v16bf b1 = ldfrag16(kb + (size_t)(16 + ln) * HDIM + c * 32 + (hi ? 8 : 0));
      s0 = __builtin_amdgcn_wmma_f32_16x16x32_bf16(false, qf[c], false, b0, (short)0, s0, false, false);
      s1 = __builtin_amdgcn_wmma_f32_16x16x32_bf16(false, qf[c], false, b1, (short)0, s1, false, false);
    }
    // scale + causal mask + online softmax (per-row reduce across the 16-lane half)
#pragma unroll
    for (int r = 0; r < 8; ++r) {
      int qi = qBase + r + (hi ? 8 : 0);
      float e0 = s0[r] * scale, e1 = s1[r] * scale;
      if (kBase + ln > qi)      e0 = -3.0e38f;
      if (kBase + 16 + ln > qi) e1 = -3.0e38f;
      float bm = fmaxf(e0, e1);
      for (int m = 8; m; m >>= 1) bm = fmaxf(bm, __shfl_xor(bm, m, 32));
      float mnew = fmaxf(mrun[r], bm);
      float alpha = __expf(mrun[r] - mnew);
      float p0 = __expf(e0 - mnew);
      float p1 = __expf(e1 - mnew);
      float ps = p0 + p1;
      for (int m = 8; m; m >>= 1) ps += __shfl_xor(ps, m, 32);
      lrun[r] = lrun[r] * alpha + ps;
      mrun[r] = mnew;
#pragma unroll
      for (int n = 0; n < 8; ++n) o[n][r] *= alpha;
      s0[r] = p0;
      s1[r] = p1;
    }
    // P (C-layout) -> LDS -> A-layout bf16 (vectorized b128 LDS reads)
#pragma unroll
    for (int r = 0; r < 8; ++r) {
      int m = r + (hi ? 8 : 0);
      pbuf[m * 32 + ln]      = s0[r];
      pbuf[m * 32 + 16 + ln] = s1[r];
    }
    __syncthreads();
    v16bf pa;
    {
      int ka = hi ? 8 : 0;
      v4f f0 = *(const v4f*)&pbuf[ln * 32 + ka];
      v4f f1 = *(const v4f*)&pbuf[ln * 32 + ka + 4];
      v4f f2 = *(const v4f*)&pbuf[ln * 32 + ka + 16];
      v4f f3 = *(const v4f*)&pbuf[ln * 32 + ka + 20];
#pragma unroll
      for (int e = 0; e < 4; ++e) {
        pa[e]      = f2bf(f0[e]);
        pa[e + 4]  = f2bf(f1[e]);
        pa[e + 8]  = f2bf(f2[e]);
        pa[e + 12] = f2bf(f3[e]);
      }
    }
    __syncthreads();
    // O += P @ V : 8 output n-tiles of 16 (contiguous b128 loads from LDS V^T tile)
#pragma unroll
    for (int n = 0; n < 8; ++n) {
      v16bf vfr = ldfrag16(vb + (size_t)(n * 16 + ln) * 32 + (hi ? 8 : 0));
      o[n] = __builtin_amdgcn_wmma_f32_16x16x32_bf16(false, pa, false, vfr, (short)0, o[n], false, false);
    }
  }

#pragma unroll
  for (int r = 0; r < 8; ++r) {
    float inv = 1.0f / lrun[r];
    int row = qBase + r + (hi ? 8 : 0);
    float* yr = Y + ((size_t)b * SEQ + row) * DIMN + (size_t)h * HDIM;
#pragma unroll
    for (int n = 0; n < 8; ++n) yr[n * 16 + ln] = o[n][r] * inv;
  }
}

// ---------------- y - (y . v_hat) v_hat  -> bf16 ----------------
__global__ void vproj(const float* __restrict__ Yf, const float* __restrict__ Vf,
                      __bf16* __restrict__ Yout) {
  int t = blockIdx.x, kh = blockIdx.y, b = blockIdx.z;
  int d = threadIdx.x;  // 0..127
  const float* vp = Vf + ((size_t)b * SEQ + t) * KVD + kh * HDIM;
  __shared__ float red[4];
  float vv = vp[d];
  float ss = vv * vv;
  for (int m = 16; m; m >>= 1) ss += __shfl_xor(ss, m, 32);
  if ((d & 31) == 0) red[d >> 5] = ss;
  __syncthreads();
  float norm = sqrtf(red[0] + red[1] + red[2] + red[3]);
  float vn = vv / fmaxf(norm, 1e-12f);
  for (int g = 0; g < GRP; ++g) {
    int hh = kh * GRP + g;
    const float* yp = Yf + ((size_t)b * SEQ + t) * DIMN + hh * HDIM;
    float y = yp[d];
    float dp = y * vn;
    for (int m = 16; m; m >>= 1) dp += __shfl_xor(dp, m, 32);
    __syncthreads();
    if ((d & 31) == 0) red[d >> 5] = dp;
    __syncthreads();
    float dot = red[0] + red[1] + red[2] + red[3];
    Yout[((size_t)b * SEQ + t) * DIMN + hh * HDIM + d] = f2bf(y - dot * vn);
  }
}

extern "C" void kernel_launch(void* const* d_in, const int* in_sizes, int n_in,
                              void* d_out, int out_size, void* d_ws, size_t ws_size,
                              hipStream_t stream) {
  (void)in_sizes; (void)n_in; (void)out_size; (void)ws_size;
  const float* x  = (const float*)d_in[0];
  const float* Wq = (const float*)d_in[1];
  const float* Wk = (const float*)d_in[2];
  const float* Wv = (const float*)d_in[3];
  const float* Wp = (const float*)d_in[4];
  const float* qg = (const float*)d_in[5];
  const float* qA = (const float*)d_in[6];
  const float* qB = (const float*)d_in[7];
  const float* kA = (const float*)d_in[8];
  const float* kB = (const float*)d_in[9];
  const float* vA = (const float*)d_in[10];
  const float* vB = (const float*)d_in[11];
  const float* pA = (const float*)d_in[12];
  const float* pB = (const float*)d_in[13];

  char* ws = (char*)d_ws;
  size_t off = 0;
  auto take = [&](size_t bytes) -> char* {
    char* p = ws + off;
    off += (bytes + 255) & ~(size_t)255;
    return p;
  };
  const size_t MT = (size_t)BATCH * SEQ;  // 4096 rows
  __bf16* Xbf = (__bf16*)take(MT * DIMN * 2);
  __bf16* Wqe = (__bf16*)take((size_t)DIMN * DIMN * 2);  // transposed (N x K)
  __bf16* Wke = (__bf16*)take((size_t)DIMN * KVD * 2);
  __bf16* Wve = (__bf16*)take((size_t)DIMN * KVD * 2);
  __bf16* Wpe = (__bf16*)take((size_t)DIMN * DIMN * 2);
  float*  Qf  = (float*)take(MT * DIMN * 4);
  float*  Kf  = (float*)take(MT * KVD * 4);
  float*  Vf  = (float*)take(MT * KVD * 4);
  __bf16* Qbf = (__bf16*)take(MT * DIMN * 2);
  __bf16* Kbf = (__bf16*)take(MT * KVD * 2);
  __bf16* Vtb = (__bf16*)take(MT * KVD * 2);  // transposed (b,kh,hd,t)
  float*  Yf  = (float*)take(MT * DIMN * 4);
  __bf16* Ybf = (__bf16*)take(MT * DIMN * 2);

  // 1) fold LoRA into effective bf16 weights, stored transposed
  build_weight_t<<<(DIMN * DIMN + 255) / 256, 256, 0, stream>>>(Wq, qA, qB, Wqe, DIMN, DIMN);
  build_weight_t<<<(DIMN * KVD + 255) / 256, 256, 0, stream>>>(Wk, kA, kB, Wke, DIMN, KVD);
  build_weight_t<<<(DIMN * KVD + 255) / 256, 256, 0, stream>>>(Wv, vA, vB, Wve, DIMN, KVD);
  build_weight_t<<<(DIMN * DIMN + 255) / 256, 256, 0, stream>>>(Wp, pA, pB, Wpe, DIMN, DIMN);
  // 2) x -> bf16
  cvt_bf16<<<(int)((MT * DIMN + 255) / 256), 256, 0, stream>>>(x, Xbf, (int)(MT * DIMN));
  // 3) QKV projections (WMMA bf16, f32 accum); 64x32 tile per wave
  int wavesQ = (int)(MT / 64) * (DIMN / 32);  // 4096
  int wavesK = (int)(MT / 64) * (KVD / 32);   // 1024
  gemm_bf16<<<wavesQ / 8, 256, 0, stream>>>(Xbf, Wqe, Qf, (int)MT, DIMN, DIMN);
  gemm_bf16<<<wavesK / 8, 256, 0, stream>>>(Xbf, Wke, Kf, (int)MT, KVD, DIMN);
  gemm_bf16<<<wavesK / 8, 256, 0, stream>>>(Xbf, Wve, Vf, (int)MT, KVD, DIMN);
  // 4) RMSNorm + RoPE (+gain for Q), bf16 out; V -> transposed bf16
  qk_postprocess<<<dim3(SEQ, NH, BATCH), 128, 0, stream>>>(Qf, Qbf, qg, NH);
  qk_postprocess<<<dim3(SEQ, NKV, BATCH), 128, 0, stream>>>(Kf, Kbf, nullptr, NKV);
  cvt_v_t<<<(int)((MT * KVD + 255) / 256), 256, 0, stream>>>(Vf, Vtb);
  // 5) causal flash attention (WMMA + async LDS double-buffering)
  attn_fwd<<<dim3(SEQ / 16, NH, BATCH), 32, 0, stream>>>(Qbf, Kbf, Vtb, Yf);
  // 6) remove projection along normalized v
  vproj<<<dim3(SEQ, NKV, BATCH), 128, 0, stream>>>(Yf, Vf, Ybf);
  // 7) output projection (WMMA bf16) straight into d_out (f32)
  gemm_bf16<<<wavesQ / 8, 256, 0, stream>>>(Ybf, Wpe, (float*)d_out, (int)MT, DIMN, DIMN);
}
